// AblatedPairEnergies_g_18296560681557
// MI455X (gfx1250) — matrix operand order repaired and checked by reading, not compiled, not run on hardware
//
#include <hip/hip_runtime.h>

typedef __attribute__((ext_vector_type(16))) _Float16 v16h;
typedef __attribute__((ext_vector_type(8)))  _Float16 v8h;
typedef __attribute__((ext_vector_type(8)))  float    v8f;

#define LL    1024
#define KNB   30
#define HH    128
#define OO    400
#define CIN   384          // 3*H
#define NT    25           // O/16
#define WH_ROWS 448        // 28 tiles * 16, padded so every wave runs 7 tiles w/o divergence
#define ROWS_PB 32         // M rows per block (2 M-tiles per wave -> 2x B reuse)

// ---------------------------------------------------------------------------
// Kernel 1: masked pairwise distances + top-30 (stable argmin, lowest index on tie)
// ---------------------------------------------------------------------------
__device__ __forceinline__ void amin(float& bv, int& bi, float v, int i) {
    if (v < bv || (v == bv && i < bi)) { bv = v; bi = i; }
}

__global__ void __launch_bounds__(256) knn_topk_kernel(const float* __restrict__ X,
                                                       const float* __restrict__ mask,
                                                       int* __restrict__ Eidx) {
    const int row = blockIdx.x;          // b*L + i
    const int b   = row >> 10;
    const int i   = row & (LL - 1);
    const int tid = threadIdx.x;

    __shared__ float sxi[3];
    __shared__ float wv[8];
    __shared__ int   wi[8];
    __shared__ int   swin;
    __shared__ float sdmax;

    if (tid < 3) sxi[tid] = X[((size_t)(b * LL + i) * 4 + 1) * 3 + tid];
    __syncthreads();
    const float xi = sxi[0], yi = sxi[1], zi = sxi[2];
    const float mi = mask[b * LL + i];

    float dv[4], m2v[4];
#pragma unroll
    for (int t = 0; t < 4; ++t) {
        const int j = tid + (t << 8);
        const float* p = X + ((size_t)(b * LL + j) * 4 + 1) * 3;
        const float dx = p[0] - xi, dy = p[1] - yi, dz = p[2] - zi;
        const float m2 = mi * mask[b * LL + j];
        m2v[t] = m2;
        dv[t]  = m2 * sqrtf(dx * dx + dy * dy + dz * dz + 1e-6f);
    }

    // row max of masked D (for the (1-m2)*D_max adjustment)
    float lm = fmaxf(fmaxf(dv[0], dv[1]), fmaxf(dv[2], dv[3]));
#pragma unroll
    for (int off = 16; off; off >>= 1) lm = fmaxf(lm, __shfl_down(lm, off));
    if ((tid & 31) == 0) wv[tid >> 5] = lm;
    __syncthreads();
    if (tid == 0) {
        float m = wv[0];
        for (int t = 1; t < 8; ++t) m = fmaxf(m, wv[t]);
        sdmax = m;
    }
    __syncthreads();
    const float dmax = sdmax;
#pragma unroll
    for (int t = 0; t < 4; ++t) dv[t] = dv[t] + (1.0f - m2v[t]) * dmax;

    for (int k = 0; k < KNB; ++k) {
        float bv = 3.4e38f;
        int   bi = 0x7fffffff;
#pragma unroll
        for (int t = 0; t < 4; ++t) amin(bv, bi, dv[t], tid + (t << 8));
#pragma unroll
        for (int off = 16; off; off >>= 1) {
            const float ov = __shfl_down(bv, off);
            const int   oi = __shfl_down(bi, off);
            amin(bv, bi, ov, oi);
        }
        if ((tid & 31) == 0) { wv[tid >> 5] = bv; wi[tid >> 5] = bi; }
        __syncthreads();
        if (tid == 0) {
            float cv = wv[0]; int ci = wi[0];
            for (int t = 1; t < 8; ++t) amin(cv, ci, wv[t], wi[t]);
            swin = ci;
            Eidx[row * KNB + k] = ci;
        }
        __syncthreads();
        const int w = swin;
        if ((w & 255) == tid) dv[w >> 8] = 3.4e38f;   // knock out winner
        __syncthreads();
    }
}

// ---------------------------------------------------------------------------
// Kernel 2: W_w (O x 3H, f32, row-major) -> f16, zero-padded to 448 rows
// ---------------------------------------------------------------------------
__global__ void convert_w_kernel(const float* __restrict__ W, _Float16* __restrict__ Wh) {
    const int idx = blockIdx.x * 256 + threadIdx.x;
    if (idx >= WH_ROWS * CIN) return;
    const int n = idx / CIN;
    Wh[idx] = (n < OO) ? (_Float16)W[idx] : (_Float16)0.0f;
}

// ---------------------------------------------------------------------------
// Kernel 3: gather h_E tile into LDS (f16), WMMA GEMM vs W^T, add bias.
// Block = 128 threads (4 waves), 32 M rows. Each wave: 2 M-tiles x 7 N-tiles,
// so every B fragment feeds two v_wmma (halves L2 traffic on W).
// ---------------------------------------------------------------------------
#define A_STRIDE 392   // 384 + 8-half pad (16B) to stagger LDS banks

__global__ void __launch_bounds__(128) pair_energy_gemm_kernel(
    const float* __restrict__ V, const float* __restrict__ E,
    const _Float16* __restrict__ Wh, const float* __restrict__ Wb,
    const int* __restrict__ Eidx, float* __restrict__ out) {

    __shared__ __align__(32) _Float16 Ah[ROWS_PB * A_STRIDE];
    __shared__ int s_b[ROWS_PB], s_i[ROWS_PB], s_e0[ROWS_PB], s_ej[ROWS_PB];

    const int tid      = threadIdx.x;
    const int mt       = blockIdx.x;            // 32-row M block
    const int row_base = mt * ROWS_PB;

    if (tid < ROWS_PB) {
        const int gm  = row_base + tid;
        const int b   = gm / (LL * KNB);
        const int rem = gm - b * (LL * KNB);
        const int i   = rem / KNB;
        const int kn  = rem - i * KNB;
        const int eb  = (b * LL + i) * KNB;
        s_b[tid]  = b;
        s_i[tid]  = i;
        s_e0[tid] = Eidx[eb];         // nearest neighbor (self) -> h_i
        s_ej[tid] = Eidx[eb + kn];    // k-th neighbor -> h_j, E_nb
    }
    __syncthreads();

    // Stage A tile: 32 rows x 384 cols, gathered + converted to f16
    for (int idx = tid; idx < ROWS_PB * CIN; idx += 128) {
        const int r = idx / CIN;
        const int c = idx - r * CIN;
        const int b = s_b[r];
        float v;
        if (c < HH)            v = V[(size_t)(b * LL + s_e0[r]) * HH + c];
        else if (c < 2 * HH)   v = V[(size_t)(b * LL + s_ej[r]) * HH + (c - HH)];
        else                   v = E[(((size_t)(b * LL + s_i[r])) * LL + s_ej[r]) * HH + (c - 2 * HH)];
        Ah[r * A_STRIDE + c] = (_Float16)v;
    }
    __syncthreads();

    const int lane = tid & 31;
    const int w    = tid >> 5;       // wave id 0..3, uniform per wave
    const int lhi  = lane >> 4;      // 0/1
    const int nlo  = lane & 15;
    const int m    = nlo;            // A-fragment row within each 16-row M-tile

    v8f acc0[7] = {};                // M-tile 0 (rows 0..15)
    v8f acc1[7] = {};                // M-tile 1 (rows 16..31)

    for (int kt = 0; kt < 12; ++kt) {
        // A fragments: halves e=0..7 -> k=kb..kb+7 ; e=8..15 -> k=kb+8..kb+15
        const int kb = kt * 32 + lhi * 16;
        const v8h a0l = *(const v8h*)(&Ah[m * A_STRIDE + kb]);
        const v8h a0h = *(const v8h*)(&Ah[m * A_STRIDE + kb + 8]);
        const v8h a1l = *(const v8h*)(&Ah[(16 + m) * A_STRIDE + kb]);
        const v8h a1h = *(const v8h*)(&Ah[(16 + m) * A_STRIDE + kb + 8]);
        v16h a0, a1;
#pragma unroll
        for (int e = 0; e < 8; ++e) {
            a0[e] = a0l[e]; a0[e + 8] = a0h[e];
            a1[e] = a1l[e]; a1[e + 8] = a1h[e];
        }

#pragma unroll
        for (int t = 0; t < 7; ++t) {
            const int n = (w * 7 + t) * 16 + nlo;
            // B fragment: lane holds column n, 16 consecutive k (32B aligned)
            const v16h bf = *(const v16h*)(Wh + (size_t)n * CIN + kt * 32 + lhi * 16);
            acc0[t] = __builtin_amdgcn_wmma_f32_16x16x32_f16(
                false, a0, false, bf, (short)0, acc0[t], false, false);
            acc1[t] = __builtin_amdgcn_wmma_f32_16x16x32_f16(
                false, a1, false, bf, (short)0, acc1[t], false, false);
        }
    }

    // Epilogue: C VGPR r -> M = r + 8*lhi, N = nlo (per lane). Bias is per-lane const.
#pragma unroll
    for (int t = 0; t < 7; ++t) {
        const int nt = w * 7 + t;
        if (nt < NT) {
            const int   n    = nt * 16 + nlo;
            const float bias = Wb[n];
#pragma unroll
            for (int r = 0; r < 8; ++r) {
                const int gm0 = row_base + r + lhi * 8;
                out[(size_t)gm0 * OO + n]        = acc0[t][r] + bias;
                out[(size_t)(gm0 + 16) * OO + n] = acc1[t][r] + bias;
            }
        }
    }
}

// ---------------------------------------------------------------------------
extern "C" void kernel_launch(void* const* d_in, const int* in_sizes, int n_in,
                              void* d_out, int out_size, void* d_ws, size_t ws_size,
                              hipStream_t stream) {
    const float* V_embed = (const float*)d_in[0];
    const float* E_embed = (const float*)d_in[1];
    const float* X       = (const float*)d_in[2];
    const float* x_mask  = (const float*)d_in[3];
    // d_in[4] chain_idx: unused by reference
    const float* W_w     = (const float*)d_in[5];
    const float* W_b     = (const float*)d_in[6];

    float* h_EV = (float*)d_out;
    int*   Eidx = (int*)d_out + (size_t)2 * LL * KNB * OO;   // tuple output tail
    _Float16* Wh = (_Float16*)d_ws;                          // 448*384*2 = 336 KB

    knn_topk_kernel<<<2 * LL, 256, 0, stream>>>(X, x_mask, Eidx);
    convert_w_kernel<<<(WH_ROWS * CIN + 255) / 256, 256, 0, stream>>>(W_w, Wh);
    pair_energy_gemm_kernel<<<(2 * LL * KNB) / ROWS_PB, 128, 0, stream>>>(
        V_embed, E_embed, Wh, W_b, Eidx, h_EV);
}
